// HopfieldPooling_44684839748141
// MI455X (gfx1250) — matrix operand order repaired and checked by reading, not compiled, not run on hardware
//
#include <hip/hip_runtime.h>
#include <hip/hip_bf16.h>

typedef unsigned int   u32;
typedef unsigned short u16;
typedef __attribute__((ext_vector_type(4)))  u32    v4u;
typedef __attribute__((ext_vector_type(8)))  float  v8f;
typedef __attribute__((ext_vector_type(16))) __bf16 v16bf;

union Frag16 { v4u q[2]; v16bf v; };

__device__ __forceinline__ v8f wmma_bf16(const Frag16& a, const Frag16& b, v8f c) {
  return __builtin_amdgcn_wmma_f32_16x16x32_bf16(false, a.v, false, b.v,
                                                 (short)0, c, false, false);
}

// CDNA5 async global->LDS copy (ASYNCcnt-tracked), 16B per lane.
__device__ __forceinline__ void async_b128(u32 lds_addr, const void* gaddr) {
  asm volatile("global_load_async_to_lds_b128 %0, %1, off"
               :: "v"(lds_addr), "v"(gaddr) : "memory");
}
__device__ __forceinline__ void wait_async_le4() {
  asm volatile("s_wait_asynccnt 4" ::: "memory");
}
__device__ __forceinline__ void wait_async_0() {
  asm volatile("s_wait_asynccnt 0" ::: "memory");
}

// ---------------------------------------------------------------------------
// Prep kernels
// ---------------------------------------------------------------------------
// query f32 -> bf16 (8 elements / thread)
__global__ void prep_q(const float* __restrict__ X, __bf16* __restrict__ Y) {
  size_t i = ((size_t)blockIdx.x * blockDim.x + threadIdx.x) * 8;
  float4 a = *(const float4*)(X + i);
  float4 b = *(const float4*)(X + i + 4);
  __bf16 o[8] = {(__bf16)a.x, (__bf16)a.y, (__bf16)a.z, (__bf16)a.w,
                 (__bf16)b.x, (__bf16)b.y, (__bf16)b.z, (__bf16)b.w};
  *(v4u*)(Y + i) = *(const v4u*)o;
}

// keys_flat[s][n] = key_param[s][:].Wk[:][n] + bk[n];  kH[h][s][e] bf16
__global__ void prep_keys(const float* __restrict__ keyp, const float* __restrict__ Wk,
                          const float* __restrict__ bk,
                          float* __restrict__ keys_flat, __bf16* __restrict__ kH) {
  int idx = blockIdx.x * blockDim.x + threadIdx.x;   // 0..32767
  int s = idx >> 10, n = idx & 1023;
  const float* kp = keyp + (size_t)s * 1024;
  float acc = bk[n];
  for (int k = 0; k < 1024; ++k) acc += kp[k] * Wk[(size_t)k * 1024 + n];
  keys_flat[idx] = acc;
  int h = n >> 6, e = n & 63;
  kH[((size_t)h * 32 + s) * 64 + e] = (__bf16)acc;
}

// values[s][n] = keys_flat[s][:].Wv[:][n] + bv[n];  vT[h][d][s] bf16
__global__ void prep_values(const float* __restrict__ keys_flat, const float* __restrict__ Wv,
                            const float* __restrict__ bv, __bf16* __restrict__ vT) {
  int idx = blockIdx.x * blockDim.x + threadIdx.x;
  int s = idx >> 10, n = idx & 1023;
  const float* kf = keys_flat + (size_t)s * 1024;
  float acc = bv[n];
  for (int k = 0; k < 1024; ++k) acc += kf[k] * Wv[(size_t)k * 1024 + n];
  int h = n >> 6, d = n & 63;
  vT[((size_t)h * 64 + d) * 32 + s] = (__bf16)acc;
}

// WqT[n][k] = bf16(Wq[k][n]); WoT[n][k] = bf16(Wo[k][n])
__global__ void prep_w(const float* __restrict__ Wq, const float* __restrict__ Wo,
                       __bf16* __restrict__ WqT, __bf16* __restrict__ WoT) {
  size_t i = (size_t)blockIdx.x * blockDim.x + threadIdx.x;  // 0..1M-1
  int k = (int)(i >> 10), n = (int)(i & 1023);
  WqT[(size_t)n * 1024 + k] = (__bf16)Wq[i];
  WoT[(size_t)n * 1024 + k] = (__bf16)Wo[i];
}

// ---------------------------------------------------------------------------
// Tiled WMMA GEMM: Y[M,1024] = Xbf16[M,1024] @ W + bias, W transposed bf16.
// 128x128 tile / 256-thread WG; 8 waves, each 32x64 (2x4 WMMA accums);
// double-buffered LDS filled by async global->LDS copies.
// ---------------------------------------------------------------------------
#define PAD 40   // 80B row stride: 16B aligned, conflict-free 16-lane reads

// 4 async instructions per wave: A tile 128x32 + B tile 128x32 (bf16)
__device__ __forceinline__ void stage_tiles(const __bf16* __restrict__ X,
                                            const __bf16* __restrict__ Wt,
                                            int row0, int col0, int kc,
                                            __bf16 (*sa)[PAD], __bf16 (*sb)[PAD],
                                            int tid) {
#pragma unroll
  for (int t = 0; t < 2; ++t) {
    int idx = tid + t * 256;          // 0..511 b128 slots
    int r  = idx >> 2;                // 0..127
    int k8 = (idx & 3) << 3;          // 0,8,16,24 halves
    async_b128((u32)(size_t)&sa[r][k8], X  + (size_t)(row0 + r) * 1024 + kc + k8);
    async_b128((u32)(size_t)&sb[r][k8], Wt + (size_t)(col0 + r) * 1024 + kc + k8);
  }
}

template <bool OUT_F32>
__global__ void __launch_bounds__(256)
gemm_bf16_kernel(const __bf16* __restrict__ X, const __bf16* __restrict__ Wt,
                 const float* __restrict__ bias, void* __restrict__ Yv) {
  __shared__ __bf16 SA[2][128][PAD];
  __shared__ __bf16 SB[2][128][PAD];
  const int N = 1024;
  const int tid  = threadIdx.x;
  const int lane = tid & 31;
  const int wave = tid >> 5;           // 0..7
  const int wm = wave >> 1;            // 0..3  (32-row group)
  const int wn = wave & 1;             // 0..1  (64-col group)
  const int row0 = blockIdx.y * 128;
  const int col0 = blockIdx.x * 128;
  const int nl = lane & 15;
  const int hi = (lane >> 4) & 1;

  v8f acc[2][4] = {};

  stage_tiles(X, Wt, row0, col0, 0, SA[0], SB[0], tid);

  int buf = 0;
  for (int kc = 0; kc < 1024; kc += 32, buf ^= 1) {
    if (kc + 32 < 1024) {
      stage_tiles(X, Wt, row0, col0, kc + 32, SA[buf ^ 1], SB[buf ^ 1], tid);
      wait_async_le4();                // current chunk landed; next in flight
    } else {
      wait_async_0();
    }
    __syncthreads();

    Frag16 a[2], b[4];
    {
      const int k0 = hi ? 16 : 0;      // A: 16 consecutive K per lane
      const int kb = hi ? 16 : 0;      // B: 16 consecutive K per lane
#pragma unroll
      for (int i = 0; i < 2; ++i) {
        int r = wm * 32 + i * 16 + nl;
        a[i].q[0] = *(const v4u*)&SA[buf][r][k0];
        a[i].q[1] = *(const v4u*)&SA[buf][r][k0 + 8];
      }
#pragma unroll
      for (int j = 0; j < 4; ++j) {
        int n = wn * 64 + j * 16 + nl;
        b[j].q[0] = *(const v4u*)&SB[buf][n][kb];
        b[j].q[1] = *(const v4u*)&SB[buf][n][kb + 8];
      }
    }
#pragma unroll
    for (int i = 0; i < 2; ++i)
#pragma unroll
      for (int j = 0; j < 4; ++j)
        acc[i][j] = wmma_bf16(a[i], b[j], acc[i][j]);

    __syncthreads();                   // all reads done before buf is reused
  }

#pragma unroll
  for (int i = 0; i < 2; ++i)
#pragma unroll
    for (int j = 0; j < 4; ++j) {
      int col = col0 + wn * 64 + j * 16 + nl;
      float bb = bias[col];
#pragma unroll
      for (int r = 0; r < 8; ++r) {
        int row = row0 + wm * 32 + i * 16 + hi * 8 + r;
        float val = acc[i][j][r] + bb;
        if (OUT_F32) ((float*)Yv)[(size_t)row * N + col]  = val;
        else         ((__bf16*)Yv)[(size_t)row * N + col] = (__bf16)val;
      }
    }
}

// ---------------------------------------------------------------------------
// Fused attention: scores (WMMA) -> sparsemax (exact, sort-32) -> ctx (WMMA)
// 64 rows per WG (4 waves x 16 rows), loop over 16 heads.
// ctx is stored with the reference's "mix" permutation applied.
// ---------------------------------------------------------------------------
__global__ void __launch_bounds__(128)
attn_kernel(const __bf16* __restrict__ Q, const __bf16* __restrict__ kH,
            const __bf16* __restrict__ vT, __bf16* __restrict__ ctx) {
  __shared__ float  scoreT[64][33];
  __shared__ __bf16 probsT[64][PAD];

  const int tid  = threadIdx.x;
  const int lane = tid & 31;
  const int wave = tid >> 5;
  const int row0  = blockIdx.x * 64;
  const int rbase = wave * 16;
  const int nl = lane & 15;
  const int hi = (lane >> 4) & 1;

  for (int h = 0; h < 16; ++h) {
    // ---- scores[16 rows, 32 patterns] = Q_h @ K_h^T, K-dim 64 (2 chunks)
    v8f sc[2] = {};
#pragma unroll
    for (int kc = 0; kc < 2; ++kc) {
      Frag16 a;
      {
        int row = row0 + rbase + nl;
        const __bf16* p = Q + (size_t)row * 1024 + h * 64 + kc * 32 + (hi ? 16 : 0);
        a.q[0] = *(const v4u*)p;
        a.q[1] = *(const v4u*)(p + 8);
      }
#pragma unroll
      for (int nt = 0; nt < 2; ++nt) {
        Frag16 b;
        int s = nt * 16 + nl;
        const __bf16* p = kH + ((size_t)h * 32 + s) * 64 + kc * 32 + (hi ? 16 : 0);
        b.q[0] = *(const v4u*)p;
        b.q[1] = *(const v4u*)(p + 8);
        sc[nt] = wmma_bf16(a, b, sc[nt]);
      }
    }
    // spill scaled scores to LDS (scale = 1/sqrt(64))
#pragma unroll
    for (int nt = 0; nt < 2; ++nt)
#pragma unroll
      for (int r = 0; r < 8; ++r)
        scoreT[rbase + hi * 8 + r][nt * 16 + nl] = sc[nt][r] * 0.125f;
    __syncthreads();

    // ---- exact sparsemax over S=32, one row per thread
    if (tid < 64) {
      float z[32], zs[32];
      for (int t = 0; t < 32; ++t) { z[t] = scoreT[tid][t]; zs[t] = z[t]; }
      for (int i = 1; i < 32; ++i) {            // insertion sort, descending
        float key = zs[i]; int jj = i - 1;
        while (jj >= 0 && zs[jj] < key) { zs[jj + 1] = zs[jj]; --jj; }
        zs[jj + 1] = key;
      }
      float cum = 0.0f, cumk = 0.0f, kf = 1.0f;
      for (int i = 0; i < 32; ++i) {
        cum += zs[i];
        if (1.0f + (float)(i + 1) * zs[i] > cum) { kf = (float)(i + 1); cumk = cum; }
      }
      float tau = (cumk - 1.0f) / kf;
      for (int t = 0; t < 32; ++t)
        probsT[tid][t] = (__bf16)fmaxf(z[t] - tau, 0.0f);
    }
    __syncthreads();

    // ---- ctx[16 rows, 64] = probs @ V_h (K=32, 4 N-tiles), permuted store
    Frag16 a;
    {
      int row = rbase + nl;
      const __bf16* p = &probsT[row][hi ? 16 : 0];
      a.q[0] = *(const v4u*)p;
      a.q[1] = *(const v4u*)(p + 8);
    }
#pragma unroll
    for (int nt = 0; nt < 4; ++nt) {
      Frag16 b;
      int d0 = nt * 16 + nl;
      const __bf16* p = vT + ((size_t)h * 64 + d0) * 32 + (hi ? 16 : 0);
      b.q[0] = *(const v4u*)p;
      b.q[1] = *(const v4u*)(p + 8);
      v8f cx = {};
      cx = wmma_bf16(a, b, cx);
#pragma unroll
      for (int r = 0; r < 8; ++r) {
        int rowg = row0 + rbase + hi * 8 + r;          // b*4096 + l
        int b_   = rowg >> 12;
        int l    = rowg & 4095;
        // mix view: dest row = b*4096 + h*256 + l/16, dest col = (l%16)*64 + d
        size_t drow = ((size_t)b_ << 12) + h * 256 + (l >> 4);
        int    dcol = (l & 15) * 64 + d0;
        ctx[drow * 1024 + dcol] = (__bf16)cx[r];
      }
    }
    __syncthreads();
  }
}

// ---------------------------------------------------------------------------
extern "C" void kernel_launch(void* const* d_in, const int* in_sizes, int n_in,
                              void* d_out, int out_size, void* d_ws, size_t ws_size,
                              hipStream_t stream) {
  const float* query     = (const float*)d_in[0];
  const float* key_param = (const float*)d_in[1];
  const float* Wq        = (const float*)d_in[2];
  const float* bq        = (const float*)d_in[3];
  const float* Wk        = (const float*)d_in[4];
  const float* bk        = (const float*)d_in[5];
  const float* Wv        = (const float*)d_in[6];
  const float* bv        = (const float*)d_in[7];
  const float* Wo        = (const float*)d_in[8];
  const float* bo        = (const float*)d_in[9];

  char* ws = (char*)d_ws;
  size_t off = 0;
  auto alloc = [&](size_t bytes) -> void* {
    void* p = ws + off;
    off = (off + bytes + 255) & ~(size_t)255;
    return p;
  };
  float*  keys_flat = (float*)alloc((size_t)32 * 1024 * 4);
  __bf16* kH        = (__bf16*)alloc((size_t)16 * 32 * 64 * 2);
  __bf16* vT        = (__bf16*)alloc((size_t)16 * 64 * 32 * 2);
  __bf16* WqT       = (__bf16*)alloc((size_t)1024 * 1024 * 2);
  __bf16* WoT       = (__bf16*)alloc((size_t)1024 * 1024 * 2);
  __bf16* Qx        = (__bf16*)alloc((size_t)32768 * 1024 * 2);  // query bf16
  __bf16* Qb        = (__bf16*)alloc((size_t)32768 * 1024 * 2);  // Q proj bf16
  __bf16* ctx       = (__bf16*)alloc((size_t)32768 * 1024 * 2);  // mixed ctx

  prep_q     <<<16384, 256, 0, stream>>>(query, Qx);
  prep_keys  <<<128,   256, 0, stream>>>(key_param, Wk, bk, keys_flat, kH);
  prep_values<<<128,   256, 0, stream>>>(keys_flat, Wv, bv, vT);
  prep_w     <<<4096,  256, 0, stream>>>(Wq, Wo, WqT, WoT);

  gemm_bf16_kernel<false><<<dim3(8, 256), 256, 0, stream>>>(Qx, WqT, bq, Qb);
  attn_kernel<<<512, 128, 0, stream>>>(Qb, kH, vT, ctx);
  gemm_bf16_kernel<true><<<dim3(8, 256), 256, 0, stream>>>(ctx, WoT, bo, (float*)d_out);
}
